// Encoder_4887672783325
// MI455X (gfx1250) — matrix operand order repaired and checked by reading, not compiled, not run on hardware
//
#include <hip/hip_runtime.h>
#include <hip/hip_bf16.h>

typedef __attribute__((ext_vector_type(16))) _Float16 v16h;
typedef __attribute__((ext_vector_type(8)))  _Float16 v8h;
typedef __attribute__((ext_vector_type(8)))  float    v8f;

#define DEV __device__ __forceinline__

DEV float sigmoidf_(float x){ return 1.f/(1.f+expf(-x)); }

// ---------------------------------------------------------------------------
// WMMA 16x16x32 f16->f32 fragment loaders (contiguous k order k = 16*hi + e, same in A and packed B).
// A fragment per lane = two contiguous 16B runs:
//   A[m][16*hi .. 16*hi+7]  and  A[m][16*hi+8 .. 16*hi+15]
// Packed-B: weight pre-swizzled so lane's 16 halves are contiguous (32B).
// ---------------------------------------------------------------------------
DEV v16h load_afrag(const _Float16* A, int lda){
  int lane = threadIdx.x & 31;
  int m = lane & 15, hi = lane >> 4;
  v8h a0 = *(const v8h*)(A + (size_t)m*lda + 16*hi);
  v8h a1 = *(const v8h*)(A + (size_t)m*lda + 16*hi + 8);
  return __builtin_shufflevector(a0, a1, 0,1,2,3,4,5,6,7,8,9,10,11,12,13,14,15);
}
DEV v16h load_afrag_r(const _Float16* A, int lda, int rows){
  int lane = threadIdx.x & 31;
  int m = lane & 15, hi = lane >> 4;
  v8h z = {};
  v8h a0 = (m < rows) ? *(const v8h*)(A + (size_t)m*lda + 16*hi)     : z;
  v8h a1 = (m < rows) ? *(const v8h*)(A + (size_t)m*lda + 16*hi + 8) : z;
  return __builtin_shufflevector(a0, a1, 0,1,2,3,4,5,6,7,8,9,10,11,12,13,14,15);
}
DEV v8f wmma_pk(const _Float16* A, int lda, const _Float16* Bpk, v8f acc){
  v16h a = load_afrag(A, lda);
  v16h b = *(const v16h*)(Bpk + (size_t)(threadIdx.x & 31)*16);
  return __builtin_amdgcn_wmma_f32_16x16x32_f16(false, a, false, b, (short)0, acc, false, false);
}
DEV v8f wmma_pk_r(const _Float16* A, int lda, int rows, const _Float16* Bpk, v8f acc){
  v16h a = load_afrag_r(A, lda, rows);
  v16h b = *(const v16h*)(Bpk + (size_t)(threadIdx.x & 31)*16);
  return __builtin_amdgcn_wmma_f32_16x16x32_f16(false, a, false, b, (short)0, acc, false, false);
}
// B given transposed: Bt[n][k] row-major, lane's fragment = Bt[n][16*hi + 0..15]
DEV v8f wmma_bt_v(const _Float16* A, int lda, const _Float16* Bt, int ldbt, v8f acc){
  v16h a = load_afrag(A, lda);
  int lane = threadIdx.x & 31;
  int m = lane & 15, hi = lane >> 4;
  v16h b = *(const v16h*)(Bt + (size_t)m*ldbt + 16*hi);
  return __builtin_amdgcn_wmma_f32_16x16x32_f16(false, a, false, b, (short)0, acc, false, false);
}

// ---------------------------------------------------------------------------
// Prep kernels
// ---------------------------------------------------------------------------
// Pack logical B (K x N) into WMMA fragment order:
// out[((kt*(N/16)+nt)*32 + lane)*16 + e] = B[kt*32 + 2*(e>>1) + 16*(lane>>4) + (e&1)][nt*16 + (lane&15)]
// transposed==0: src is (K,N) row-major; transposed==1: src is (N,K) row-major.
__global__ void k_pack(const float* __restrict__ src, _Float16* __restrict__ out,
                       int K, int N, int transposed){
  int i = blockIdx.x*blockDim.x + threadIdx.x;
  if (i >= K*N) return;
  int e = i & 15, lane = (i >> 4) & 31, rest = i >> 9;
  int ntiles = N >> 4;
  int nt = rest % ntiles, kt = rest / ntiles;
  int hi = lane >> 4, col = lane & 15;
  int k = kt*32 + 2*(e >> 1) + 16*hi + (e & 1);
  int n = nt*16 + col;
  float v = transposed ? src[(size_t)n*K + k] : src[(size_t)k*N + n];
  out[i] = (_Float16)v;
}
__global__ void k_zero_f(float* p, int n){
  int i = blockIdx.x*blockDim.x + threadIdx.x;
  if (i < n) p[i] = 0.f;
}
__global__ void k_embed(const int* __restrict__ iv, const float* __restrict__ emb,
                        float* __restrict__ e32, _Float16* __restrict__ e16, int n){
  int i = blockIdx.x*blockDim.x + threadIdx.x;
  if (i < n){
    int w = iv[i >> 7];
    float v = emb[(size_t)w*128 + (i & 127)];
    e32[i] = v; e16[i] = (_Float16)v;
  }
}

// ---------------------------------------------------------------------------
// xg = X(512,K) @ W^T (K,1536) + bias   (per direction via blockIdx.z)
// ---------------------------------------------------------------------------
__global__ void __launch_bounds__(32) k_xg(const _Float16* __restrict__ A, int K,
                                           const _Float16* __restrict__ Wpk,
                                           const float* __restrict__ bias,
                                           float* __restrict__ out){
  int nt = blockIdx.x, mt = blockIdx.y, dir = blockIdx.z;
  const _Float16* Wd = Wpk + (size_t)dir*K*1536;
  const float*    bd = bias + dir*1536;
  float*          od = out + (size_t)dir*512*1536;
  int lane = threadIdx.x & 31, n = lane & 15, hi = lane >> 4;
  int kts = K/32;
  v8f acc = {};
  for (int kt = 0; kt < kts; ++kt){
    if (kt + 1 < kts)
      __builtin_prefetch(Wd + ((size_t)(kt+1)*96 + nt)*512, 0, 3);
    acc = wmma_pk(A + (size_t)(mt*16)*K + kt*32, K,
                  Wd + ((size_t)kt*96 + nt)*512, acc);
  }
  int col = nt*16 + n;
  float bb = bd[col];
#pragma unroll
  for (int r = 0; r < 8; ++r)
    od[(size_t)(mt*16 + r + 8*hi)*1536 + col] = acc[r] + bb;
}

// ---------------------------------------------------------------------------
// GRU recurrence: one block per direction, 32 waves; 64 sequential steps.
// h in LDS (f16, M=8 padded to 16 via zero-fill loader); hg (8x1536) LDS f32.
// ---------------------------------------------------------------------------
__global__ void __launch_bounds__(1024) k_gru(const float* __restrict__ xg,
                                              const _Float16* __restrict__ whhPk,
                                              const float* __restrict__ bhh,
                                              float* __restrict__ hstate,
                                              float* __restrict__ o32,
                                              _Float16* __restrict__ o16,
                                              int stride, int colOffPerDir, int dirOffPerDir){
  __shared__ alignas(32) _Float16 h16[8*512];
  __shared__ float hg[8*1536];
  int dir = blockIdx.x;
  const float*    xgd = xg    + (size_t)dir*512*1536;
  const _Float16* Wd  = whhPk + (size_t)dir*512*1536;
  const float*    bd  = bhh   + dir*1536;
  float*          hs  = hstate + dir*4096;
  int tid = threadIdx.x;
  int lane = tid & 31, wv = tid >> 5, n16 = lane & 15, hi = lane >> 4;

  for (int i = tid; i < 4096; i += 1024) h16[i] = (_Float16)hs[i];
  __syncthreads();

  for (int t = 0; t < 64; ++t){
    int tt = dir ? (63 - t) : t;
    // hg = h @ Whh^T + bhh
    for (int nt = wv; nt < 96; nt += 32){
      v8f acc = {};
      for (int kt = 0; kt < 16; ++kt)
        acc = wmma_pk_r(h16 + kt*32, 512, 8,
                        Wd + ((size_t)kt*96 + nt)*512, acc);
      if (hi == 0){
        int col = nt*16 + n16;
        float bb = bd[col];
#pragma unroll
        for (int r = 0; r < 8; ++r) hg[r*1536 + col] = acc[r] + bb;
      }
    }
    __syncthreads();
    // gate update
    for (int i = tid; i < 4096; i += 1024){
      int b = i >> 9, j = i & 511;
      const float* xr = xgd + (size_t)(b*64 + tt)*1536;
      float r  = sigmoidf_(xr[j]        + hg[b*1536 + j]);
      float z  = sigmoidf_(xr[512 + j]  + hg[b*1536 + 512 + j]);
      float nn = tanhf   (xr[1024 + j] + r*hg[b*1536 + 1024 + j]);
      float hold = (float)h16[i];
      float h2 = (1.f - z)*nn + z*hold;
      h16[i] = (_Float16)h2;
      size_t oidx = (size_t)dir*dirOffPerDir + (size_t)(b*64 + tt)*stride + dir*colOffPerDir + j;
      o32[oidx] = h2;
      if (o16) o16[oidx] = (_Float16)h2;
    }
    __syncthreads();
  }
  for (int i = tid; i < 4096; i += 1024) hs[i] = (float)h16[i];
}

// word_out[s] = o1f + o1b ; also init node_vec and f16 mirrors
__global__ void k_add2(const float* __restrict__ o1, float* __restrict__ wout_out,
                       float* __restrict__ w32, _Float16* __restrict__ w16,
                       float* __restrict__ nv32, _Float16* __restrict__ nv16, int s){
  int i = blockIdx.x*blockDim.x + threadIdx.x;
  if (i < 262144){
    float v = o1[i] + o1[262144 + i];
    size_t g = (size_t)s*262144 + i;
    wout_out[g] = v; w32[g] = v; w16[g] = (_Float16)v;
    nv32[g] = v; nv16[g] = (_Float16)v;
  }
}

// word_operator: rows (b, s*64+l) x Wk(128,16)
__global__ void __launch_bounds__(32) k_wordop(const _Float16* __restrict__ e16,
                                               const _Float16* __restrict__ wkpk,
                                               float* __restrict__ out){
  __shared__ alignas(32) _Float16 At[16*128];
  int mt = blockIdx.x, tid = threadIdx.x;
  v8h* At8 = (v8h*)At;
  for (int i = tid; i < 256; i += 32){
    int rr = i >> 4, k = (i & 15)*8;
    int row = mt*16 + rr;
    int b = row >> 8, rem = row & 255, s = rem >> 6, l = rem & 63;
    At8[i] = *(const v8h*)(e16 + (size_t)((s*8 + b)*64 + l)*128 + k);
  }
  __syncthreads();
  v8f acc = {};
  for (int kt = 0; kt < 4; ++kt)
    acc = wmma_pk(At + kt*32, 128, wkpk + (size_t)kt*512, acc);
  int n = tid & 15, hi = tid >> 4;
#pragma unroll
  for (int r = 0; r < 8; ++r)
    out[(size_t)(mt*16 + r + 8*hi)*16 + n] = acc[r];
}

__global__ void k_sq(const float* __restrict__ e32, float* __restrict__ sq){
  int i = blockIdx.x*blockDim.x + threadIdx.x;
  if (i < 2048){
    int b = i >> 8, n = i & 255, s = n >> 6, l = n & 63;
    const float* p = e32 + (size_t)((s*8 + b)*64 + l)*128;
    float acc = 0.f;
    for (int e = 0; e < 128; ++e){ float v = p[e]; acc += v*v; }
    sq[i] = acc;
  }
}

// gram + distance (no atomics; mean via deterministic reduction)
__global__ void __launch_bounds__(32) k_gram(const _Float16* __restrict__ e16,
                                             const float* __restrict__ sq,
                                             float* __restrict__ dist){
  __shared__ alignas(32) _Float16 At[16*128];
  __shared__ alignas(32) _Float16 Bt[16*128];
  int nt = blockIdx.x, mt = blockIdx.y, b = blockIdx.z, tid = threadIdx.x;
  v8h* At8 = (v8h*)At; v8h* Bt8 = (v8h*)Bt;
  for (int i = tid; i < 256; i += 32){
    int rr = i >> 4, k = (i & 15)*8;
    int ni = mt*16 + rr;
    At8[i] = *(const v8h*)(e16 + (size_t)(((ni >> 6)*8 + b)*64 + (ni & 63))*128 + k);
    int nj = nt*16 + rr;
    Bt8[i] = *(const v8h*)(e16 + (size_t)(((nj >> 6)*8 + b)*64 + (nj & 63))*128 + k);
  }
  __syncthreads();
  v8f acc = {};
  for (int kt = 0; kt < 4; ++kt)
    acc = wmma_bt_v(At + kt*32, 128, Bt + kt*32, 128, acc);
  int n = tid & 15, hi = tid >> 4;
#pragma unroll
  for (int r = 0; r < 8; ++r){
    int i = mt*16 + r + 8*hi, j = nt*16 + n;
    float d2 = sq[b*256 + i] + sq[b*256 + j] - 2.f*acc[r];
    dist[(size_t)b*65536 + i*256 + j] = sqrtf(fmaxf(d2, 0.f) + 1e-12f);
  }
}
__global__ void k_red1(const float* __restrict__ x, float* __restrict__ part){
  __shared__ float red[256];
  int b = blockIdx.x;
  float s = 0.f;
  for (int i = threadIdx.x; i < 2048; i += 256) s += x[(size_t)b*2048 + i];
  red[threadIdx.x] = s; __syncthreads();
  for (int k = 128; k > 0; k >>= 1){ if (threadIdx.x < k) red[threadIdx.x] += red[threadIdx.x + k]; __syncthreads(); }
  if (threadIdx.x == 0) part[b] = red[0];
}
__global__ void k_red2(const float* __restrict__ part, float* __restrict__ macc){
  __shared__ float red[256];
  red[threadIdx.x] = part[threadIdx.x]; __syncthreads();
  for (int k = 128; k > 0; k >>= 1){ if (threadIdx.x < k) red[threadIdx.x] += red[threadIdx.x + k]; __syncthreads(); }
  if (threadIdx.x == 0) macc[0] = red[0];
}
__global__ void k_wwfinal(const float* __restrict__ dist, const float* __restrict__ macc,
                          float* __restrict__ out, int n){
  int i = blockIdx.x*blockDim.x + threadIdx.x;
  if (i < n){
    float mean = macc[0] * (1.f/524288.f);
    float s = 1.f/(1.f + expf(dist[i] - mean));
    out[i] = (s < 0.5f) ? 0.f : s;
  }
}

__global__ void k_depend(const int* __restrict__ parent, float* __restrict__ out){
  int i = blockIdx.x*blockDim.x + threadIdx.x;
  if (i < 131072){
    int sb = i >> 12, rem = i & 4095, ci = rem >> 6, cj = rem & 63;
    int pi = parent[sb*64 + ci], pj = parent[sb*64 + cj];
    float v = ((pi >= 0 && pi == cj) ? 1.f : 0.f) +
              ((pj >= 0 && pj == ci) ? 1.f : 0.f) +
              ((ci == cj) ? 1.f : 0.f);
    out[i] = v;
  }
}

// tree attention: tanh(concat(qpw,node_vec)@Wa_p) into tbuf (2048x512)
__global__ void __launch_bounds__(32) k_energy(const _Float16* __restrict__ w16,
                                               const _Float16* __restrict__ nv16,
                                               const int* __restrict__ parent,
                                               const _Float16* __restrict__ wappk,
                                               float* __restrict__ tbuf){
  __shared__ alignas(32) _Float16 At[16*1024];
  int nt = blockIdx.x, mt = blockIdx.y, tid = threadIdx.x;
  v8h* At8 = (v8h*)At;
  for (int i = tid; i < 2048; i += 32){
    int rr = i >> 7, k = (i & 127)*8;
    int row = mt*16 + rr;
    const _Float16* srcp;
    if (k < 512){
      int p = parent[row];
      int pc = p < 0 ? 0 : (p > 63 ? 63 : p);
      int q = (row & ~63) + pc;
      srcp = w16 + (size_t)q*512 + k;
    } else {
      srcp = nv16 + (size_t)row*512 + (k - 512);
    }
    At8[i] = *(const v8h*)srcp;
  }
  __syncthreads();
  v8f acc = {};
  for (int kt = 0; kt < 32; ++kt){
    if (kt + 1 < 32)
      __builtin_prefetch(wappk + ((size_t)(kt+1)*32 + nt)*512, 0, 3);
    acc = wmma_pk(At + kt*32, 1024, wappk + ((size_t)kt*32 + nt)*512, acc);
  }
  int n = tid & 15, hi = tid >> 4;
#pragma unroll
  for (int r = 0; r < 8; ++r)
    tbuf[(size_t)(mt*16 + r + 8*hi)*512 + nt*16 + n] = tanhf(acc[r]);
}
__global__ void k_energy_dot(const float* __restrict__ tbuf, const float* __restrict__ vap,
                             float* __restrict__ energy){
  int row = blockIdx.x*blockDim.x + threadIdx.x;
  if (row < 2048){
    const float* t = tbuf + (size_t)row*512;
    float a = 0.f;
    for (int n = 0; n < 512; ++n) a += t[n]*vap[n];
    energy[row] = a;
  }
}

// per-(s,b): softmax-by-depth, denom, ctx (deterministic)
__global__ void k_tree(const int* __restrict__ parent, const int* __restrict__ depth,
                       const float* __restrict__ energy, const float* __restrict__ nv32,
                       float* __restrict__ ctx32, _Float16* __restrict__ ctx16,
                       float* __restrict__ denom, int* __restrict__ has, int d){
  __shared__ float eS[64], dS[64];
  __shared__ int   hS[64];
  int sb = blockIdx.x, base = sb*64, tid = threadIdx.x;
  if (tid == 0){
    float m = -3.0e38f; int any = 0;
    for (int c = 0; c < 64; ++c)
      if (depth[base + c] == d){ any = 1; float ev = energy[base + c]; if (ev > m) m = ev; }
    if (!any) m = 0.f;
    for (int p = 0; p < 64; ++p) dS[p] = 0.f;
    for (int c = 0; c < 64; ++c){
      float e = (depth[base + c] == d) ? expf(energy[base + c] - m) : 0.f;
      eS[c] = e;
      int p = parent[base + c];
      if (p >= 0) dS[p] += e;
    }
    for (int p = 0; p < 64; ++p){
      hS[p] = dS[p] > 0.f;
      denom[base + p] = dS[p];
      has[base + p] = hS[p];
    }
  }
  __syncthreads();
  for (int h = tid; h < 512; h += blockDim.x){
    for (int p = 0; p < 64; ++p){
      float v = 0.f;
      for (int c = 0; c < 64; ++c){
        if (parent[base + c] == p) v += eS[c]*nv32[(size_t)(base + c)*512 + h];
      }
      float cv = hS[p] ? v/dS[p] : 0.f;
      ctx32[(size_t)(base + p)*512 + h] = cv;
      ctx16[(size_t)(base + p)*512 + h] = (_Float16)cv;
    }
  }
}

// node_vec update: tanh(concat(ctx,word_out)@Wc_p) where has
__global__ void __launch_bounds__(32) k_update(const _Float16* __restrict__ ctx16,
                                               const _Float16* __restrict__ w16,
                                               const _Float16* __restrict__ wcppk,
                                               const int* __restrict__ has,
                                               float* __restrict__ nv32,
                                               _Float16* __restrict__ nv16){
  __shared__ alignas(32) _Float16 At[16*1024];
  int nt = blockIdx.x, mt = blockIdx.y, tid = threadIdx.x;
  v8h* At8 = (v8h*)At;
  for (int i = tid; i < 2048; i += 32){
    int rr = i >> 7, k = (i & 127)*8;
    int row = mt*16 + rr;
    const _Float16* srcp = (k < 512) ? (ctx16 + (size_t)row*512 + k)
                                     : (w16   + (size_t)row*512 + (k - 512));
    At8[i] = *(const v8h*)srcp;
  }
  __syncthreads();
  v8f acc = {};
  for (int kt = 0; kt < 32; ++kt){
    if (kt + 1 < 32)
      __builtin_prefetch(wcppk + ((size_t)(kt+1)*32 + nt)*512, 0, 3);
    acc = wmma_pk(At + kt*32, 1024, wcppk + ((size_t)kt*32 + nt)*512, acc);
  }
  int n = tid & 15, hi = tid >> 4;
#pragma unroll
  for (int r = 0; r < 8; ++r){
    int row = mt*16 + r + 8*hi;
    if (has[row]){
      float v = tanhf(acc[r]);
      int col = nt*16 + n;
      nv32[(size_t)row*512 + col] = v;
      nv16[(size_t)row*512 + col] = (_Float16)v;
    }
  }
}

// span section
__global__ void k_xsp(const float* __restrict__ nv32, const float* __restrict__ pe,
                      float* __restrict__ xsp){
  int i = blockIdx.x*blockDim.x + threadIdx.x;
  if (i < 16384){
    int b = i >> 11, s = (i >> 9) & 3, h = i & 511;
    xsp[i] = nv32[(size_t)((s*8 + b)*64)*512 + h] + pe[s*512 + h];
  }
}
__global__ void k_sc(const float* __restrict__ xsp, const float* __restrict__ was,
                     const float* __restrict__ vas, const int* __restrict__ slen,
                     float* __restrict__ sc){
  __shared__ float red[256];
  int idx = blockIdx.x;                 // b*16 + q*4 + k
  int b = idx >> 4, q = (idx >> 2) & 3, kk = idx & 3;
  const float* xq = xsp + (size_t)(b*4 + q)*512;
  const float* xk = xsp + (size_t)(b*4 + kk)*512;
  float part = 0.f;
  for (int n = threadIdx.x; n < 512; n += 256){
    float dot = 0.f;
    for (int t = 0; t < 512; ++t) dot += xq[t]*was[(size_t)t*512 + n];
    for (int t = 0; t < 512; ++t) dot += xk[t]*was[(size_t)(512 + t)*512 + n];
    part += tanhf(dot)*vas[n];
  }
  red[threadIdx.x] = part; __syncthreads();
  for (int s = 128; s > 0; s >>= 1){ if (threadIdx.x < s) red[threadIdx.x] += red[threadIdx.x + s]; __syncthreads(); }
  if (threadIdx.x == 0){
    float v = red[0];
    if (kk >= slen[b]) v = -1.0e9f;
    sc[idx] = v;
  }
}
__global__ void k_spanout(const float* __restrict__ xsp, const float* __restrict__ sc,
                          const float* __restrict__ wcs, const int* __restrict__ slen,
                          float* __restrict__ out){
  __shared__ float attn[4];
  __shared__ float ctx[512];
  int idx = blockIdx.x;                 // b*4 + q
  int b = idx >> 2, q = idx & 3;
  if (threadIdx.x == 0){
    const float* s = sc + idx*4;
    float m = s[0];
    for (int k = 1; k < 4; ++k) m = fmaxf(m, s[k]);
    float e0 = expf(s[0]-m), e1 = expf(s[1]-m), e2 = expf(s[2]-m), e3 = expf(s[3]-m);
    float den = e0+e1+e2+e3;
    attn[0]=e0/den; attn[1]=e1/den; attn[2]=e2/den; attn[3]=e3/den;
  }
  __syncthreads();
  for (int h = threadIdx.x; h < 512; h += blockDim.x){
    float v = 0.f;
    for (int k = 0; k < 4; ++k) v += attn[k]*xsp[(size_t)(b*4 + k)*512 + h];
    ctx[h] = v;
  }
  __syncthreads();
  const float* xq = xsp + (size_t)idx*512;
  float maskq = (q >= slen[b]) ? 0.f : 1.f;
  for (int n = threadIdx.x; n < 512; n += blockDim.x){
    float dot = 0.f;
    for (int t = 0; t < 512; ++t) dot += ctx[t]*wcs[(size_t)t*512 + n];
    for (int t = 0; t < 512; ++t) dot += xq[t]*wcs[(size_t)(512 + t)*512 + n];
    out[(size_t)idx*512 + n] = tanhf(dot)*maskq;
  }
}
__global__ void k_sph(const float* __restrict__ span_out, const int* __restrict__ slen,
                      float* __restrict__ out){
  int i = blockIdx.x*blockDim.x + threadIdx.x;
  if (i < 4096){
    int b = i >> 9, h = i & 511;
    int q = slen[b] - 1; if (q < 0) q = 0; if (q > 3) q = 3;
    out[i] = span_out[(size_t)(b*4 + q)*512 + h];
  }
}

// ---------------------------------------------------------------------------
extern "C" void kernel_launch(void* const* d_in, const int* in_sizes, int n_in,
                              void* d_out, int out_size, void* d_ws, size_t ws_size,
                              hipStream_t stream){
  (void)in_sizes; (void)n_in; (void)out_size; (void)ws_size;
  const int*   input_var = (const int*)d_in[0];
  const int*   span_len  = (const int*)d_in[2];
  const int*   parent    = (const int*)d_in[3];
  const int*   depth     = (const int*)d_in[4];
  const float* emb  = (const float*)d_in[5];
  const float* wih0 = (const float*)d_in[6];
  const float* whh0 = (const float*)d_in[7];
  const float* bih0 = (const float*)d_in[8];
  const float* bhh0 = (const float*)d_in[9];
  const float* wih1 = (const float*)d_in[10];
  const float* whh1 = (const float*)d_in[11];
  const float* bih1 = (const float*)d_in[12];
  const float* bhh1 = (const float*)d_in[13];
  const float* pe   = (const float*)d_in[14];
  const float* Wk   = (const float*)d_in[15];
  const float* Wa_s = (const float*)d_in[16];
  const float* va_s = (const float*)d_in[17];
  const float* Wc_s = (const float*)d_in[18];
  const float* Wa_p = (const float*)d_in[19];
  const float* va_p = (const float*)d_in[20];
  const float* Wc_p = (const float*)d_in[21];

  float* out   = (float*)d_out;
  float* o_span = out;                       // (B,S,H)        16384
  float* o_wout = out + 16384;               // (S,B,L,H)      1048576
  float* o_wop  = o_wout + 1048576;          // (B,256,16)     32768
  float* o_ww   = o_wop + 32768;             // (B,256,256)    524288
  float* o_dep  = o_ww + 524288;             // (S,B,L,L)      131072
  float* o_sph  = o_dep + 131072;            // (1,B,H)        4096

  char* ws = (char*)d_ws;
  size_t off = 0;
  auto take = [&](size_t bytes)->char*{
    char* p = ws + off;
    off = (off + bytes + 255) & ~((size_t)255);
    return p;
  };
  float*    emb32  = (float*)take(262144*4);
  _Float16* emb16  = (_Float16*)take(262144*2);
  float*    xg0    = (float*)take((size_t)2*512*1536*4);
  float*    xg1    = (float*)take((size_t)2*512*1536*4);
  float*    x1_32  = (float*)take((size_t)512*1024*4);
  _Float16* x1_16  = (_Float16*)take((size_t)512*1024*2);
  float*    o1     = (float*)take((size_t)2*512*512*4);
  float*    hstate = (float*)take(16384*4);
  float*    wout32 = (float*)take((size_t)1048576*4);
  _Float16* wout16 = (_Float16*)take((size_t)1048576*2);
  float*    nv32   = (float*)take((size_t)1048576*4);
  _Float16* nv16   = (_Float16*)take((size_t)1048576*2);
  float*    ctx32  = (float*)take((size_t)1048576*4);
  _Float16* ctx16  = (_Float16*)take((size_t)1048576*2);
  float*    tbuf   = (float*)take((size_t)1048576*4);
  float*    energy = (float*)take(2048*4);
  float*    denomb = (float*)take(2048*4);
  int*      hasb   = (int*)take(2048*4);
  float*    distb  = (float*)take((size_t)524288*4);
  float*    sqb    = (float*)take(2048*4);
  float*    partb  = (float*)take(256*4);
  float*    maccb  = (float*)take(256);
  float*    xspb   = (float*)take(16384*4);
  float*    scb    = (float*)take(128*4);
  _Float16* wih0pk = (_Float16*)take((size_t)2*128*1536*2);
  _Float16* whh0pk = (_Float16*)take((size_t)2*512*1536*2);
  _Float16* wih1pk = (_Float16*)take((size_t)2*1024*1536*2);
  _Float16* whh1pk = (_Float16*)take((size_t)2*512*1536*2);
  _Float16* wkpk   = (_Float16*)take(2048*2);
  _Float16* wappk  = (_Float16*)take((size_t)524288*2);
  _Float16* wcppk  = (_Float16*)take((size_t)524288*2);

  // ---- weight prep (pack into WMMA fragment order) ----
  for (int d = 0; d < 2; ++d){
    k_pack<<<(128*1536 +255)/256,256,0,stream>>>(wih0 + (size_t)d*1536*128,  wih0pk + (size_t)d*128*1536,  128,  1536, 1);
    k_pack<<<(512*1536 +255)/256,256,0,stream>>>(whh0 + (size_t)d*1536*512,  whh0pk + (size_t)d*512*1536,  512,  1536, 1);
    k_pack<<<(1024*1536+255)/256,256,0,stream>>>(wih1 + (size_t)d*1536*1024, wih1pk + (size_t)d*1024*1536, 1024, 1536, 1);
    k_pack<<<(512*1536 +255)/256,256,0,stream>>>(whh1 + (size_t)d*1536*512,  whh1pk + (size_t)d*512*1536,  512,  1536, 1);
  }
  k_pack<<<(2048  +255)/256,256,0,stream>>>(Wk,   wkpk,  128,  16,  0);
  k_pack<<<(524288+255)/256,256,0,stream>>>(Wa_p, wappk, 1024, 512, 0);
  k_pack<<<(524288+255)/256,256,0,stream>>>(Wc_p, wcppk, 1024, 512, 0);
  k_embed<<<(262144+255)/256,256,0,stream>>>(input_var, emb, emb32, emb16, 262144);
  k_zero_f<<<(16384+255)/256,256,0,stream>>>(hstate, 16384);

  // ---- GRU stack over 4 segments ----
  for (int s = 0; s < 4; ++s){
    k_xg<<<dim3(96,32,2),32,0,stream>>>(emb16 + (size_t)s*512*128, 128, wih0pk, bih0, xg0);
    k_gru<<<2,1024,0,stream>>>(xg0, whh0pk, bhh0, hstate, x1_32, x1_16, 1024, 512, 0);
    k_xg<<<dim3(96,32,2),32,0,stream>>>(x1_16, 1024, wih1pk, bih1, xg1);
    k_gru<<<2,1024,0,stream>>>(xg1, whh1pk, bhh1, hstate + 2*4096, o1, (_Float16*)nullptr, 512, 0, 262144);
    k_add2<<<(262144+255)/256,256,0,stream>>>(o1, o_wout, wout32, wout16, nv32, nv16, s);
  }

  // ---- word_operator ----
  k_wordop<<<128,32,0,stream>>>(emb16, wkpk, o_wop);

  // ---- word_word ----
  k_sq<<<8,256,0,stream>>>(emb32, sqb);
  k_gram<<<dim3(16,16,8),32,0,stream>>>(emb16, sqb, distb);
  k_red1<<<256,256,0,stream>>>(distb, partb);
  k_red2<<<1,256,0,stream>>>(partb, maccb);
  k_wwfinal<<<(524288+255)/256,256,0,stream>>>(distb, maccb, o_ww, 524288);

  // ---- depend_relation ----
  k_depend<<<(131072+255)/256,256,0,stream>>>(parent, o_dep);

  // ---- tree attention loop ----
  for (int d = 10; d >= 1; --d){
    k_energy<<<dim3(32,128),32,0,stream>>>(wout16, nv16, parent, wappk, tbuf);
    k_energy_dot<<<8,256,0,stream>>>(tbuf, va_p, energy);
    k_tree<<<32,256,0,stream>>>(parent, depth, energy, nv32, ctx32, ctx16, denomb, hasb, d);
    k_update<<<dim3(32,128),32,0,stream>>>(ctx16, wout16, wcppk, hasb, nv32, nv16);
  }

  // ---- span attention ----
  k_xsp<<<(16384+255)/256,256,0,stream>>>(nv32, pe, xspb);
  k_sc<<<128,256,0,stream>>>(xspb, Wa_s, va_s, span_len, scb);
  k_spanout<<<32,256,0,stream>>>(xspb, scb, Wc_s, span_len, o_span);
  k_sph<<<(4096+255)/256,256,0,stream>>>(o_span, span_len, o_sph);
}
